// LSTM_9010841387868
// MI455X (gfx1250) — matrix operand order repaired and checked by reading, not compiled, not run on hardware
//
#include <hip/hip_runtime.h>
#include <hip/hip_bf16.h>
#include <stdint.h>

#ifndef __has_builtin
#define __has_builtin(x) 0
#endif

#define Bdim 256
#define Tdim 512
#define Hdim 1024
#define Cdim 128

typedef __attribute__((ext_vector_type(16))) __bf16 v16bf;
typedef __attribute__((ext_vector_type(8)))  float  v8f;
typedef __attribute__((ext_vector_type(4)))  int    i4;

// Pointer-to-16B-vector types with explicit address spaces, matching the
// async-LDS builtin's parameter types ('int __vector(4) *').
typedef __attribute__((address_space(1))) i4* gi4_p;   // global
typedef __attribute__((address_space(3))) i4* li4_p;   // LDS

#if __has_builtin(__builtin_amdgcn_global_load_async_to_lds_b128)
#define HAS_ASYNC_LDS 1
#else
#define HAS_ASYNC_LDS 0
#endif

// 16-byte global -> LDS copy via the CDNA5 async path (ASYNCcnt-tracked).
// OFF is the instruction's immediate offset: per the ISA it is added to BOTH
// the LDS destination and the global source address, so callers pass matching
// displacements through it and the compiler needs only one address pair.
template <int OFF>
__device__ __forceinline__ void cp_async16(void* lds, const void* g) {
#if HAS_ASYNC_LDS
  __builtin_amdgcn_global_load_async_to_lds_b128(
      (gi4_p)(uintptr_t)g,
      (li4_p)(unsigned int)(uintptr_t)lds,  // AS3 ptr == low 32 bits of flat LDS addr
      OFF, 0);
#else
  *(uint4*)((char*)lds + OFF) = *(const uint4*)((const char*)g + OFF);
#endif
}

__device__ __forceinline__ void wait_async_5() {
#if HAS_ASYNC_LDS
  asm volatile("s_wait_asynccnt 0x5" ::: "memory");
#endif
}
__device__ __forceinline__ void wait_async_0() {
#if HAS_ASYNC_LDS
  asm volatile("s_wait_asynccnt 0x0" ::: "memory");
#endif
}

__device__ __forceinline__ unsigned short f2bf(float f) {
  unsigned u = __float_as_uint(f);
  u += 0x7FFFu + ((u >> 16) & 1u);   // round-to-nearest-even
  return (unsigned short)(u >> 16);
}

__device__ __forceinline__ float fast_tanh(float x) {
#if __has_builtin(__builtin_amdgcn_tanhf)
  return __builtin_amdgcn_tanhf(x);
#else
  return tanhf(x);
#endif
}
__device__ __forceinline__ float fsigmoid(float x) {
  return 1.0f / (1.0f + __expf(-x));
}

__device__ __forceinline__ v8f wmma_bf16(v16bf a, v16bf b, v8f c) {
  // (neg_a, A, neg_b, B, c_mod, C, reuse_a, reuse_b)
  return __builtin_amdgcn_wmma_f32_16x16x32_bf16(false, a, false, b, (short)0, c,
                                                 false, false);
}

// ---------------------------------------------------------------------------
// Pack Wh (4 gate matrices [H,H] f32) into bf16 WMMA B-fragments.
// Fragment = 32 lanes x 32 bytes. k order used here (16-bit B 32x16, same as A loads):
// lane n<16 -> col n, K = 2v,2v+1 in dword v; lanes 16..31 -> col n-16,
// K = 16+2v,17+2v.
// Fragment order: [kb=K/32][ntg=N/64][gate][ntl=0..3] so each k-step's 16KB
// block slice is contiguous and maps 1:1 onto the LDS staging buffer.
// ---------------------------------------------------------------------------
__global__ void k_pack_Wh(const float* __restrict__ Wgh, const float* __restrict__ Wih,
                          const float* __restrict__ Wfh, const float* __restrict__ Woh,
                          unsigned* __restrict__ WhP) {
  unsigned d    = blockIdx.x * blockDim.x + threadIdx.x;   // dword index (2M total)
  unsigned frag = d >> 8;            // 8192 fragments
  unsigned lane = (d >> 3) & 31;
  unsigned v    = d & 7;
  unsigned kb   = frag >> 8;         // K/32
  unsigned rem  = frag & 255;
  unsigned ntg  = rem >> 4;          // N/64 group (matches blockIdx.y)
  unsigned gate = (rem >> 2) & 3;
  unsigned ntl  = rem & 3;           // 16-col tile within group
  unsigned k = kb * 32 + ((lane >> 4) << 4) + 2 * v;
  unsigned n = (ntg * 4 + ntl) * 16 + (lane & 15);
  const float* W = (gate == 0) ? Wgh : (gate == 1) ? Wih : (gate == 2) ? Wfh : Woh;
  float f0 = W[(size_t)k * Hdim + n];
  float f1 = W[(size_t)(k + 1) * Hdim + n];
  WhP[d] = (unsigned)f2bf(f0) | ((unsigned)f2bf(f1) << 16);
}

__global__ void k_pack_Wph(const float* __restrict__ Wph, unsigned* __restrict__ WphP) {
  unsigned d    = blockIdx.x * blockDim.x + threadIdx.x;   // 65536 dwords
  unsigned frag = d >> 8;
  unsigned lane = (d >> 3) & 31;
  unsigned v    = d & 7;
  unsigned kb = frag / 8, nt = frag % 8;
  unsigned k = kb * 32 + ((lane >> 4) << 4) + 2 * v;
  unsigned n = nt * 16 + (lane & 15);
  WphP[d] = (unsigned)f2bf(Wph[(size_t)k * Cdim + n]) |
            ((unsigned)f2bf(Wph[(size_t)(k + 1) * Cdim + n]) << 16);
}

__global__ void k_pack_wxb(const float* Wgx, const float* Wix, const float* Wfx,
                           const float* Wox, const float* bg, const float* bi,
                           const float* bff, const float* bo,
                           float* __restrict__ Wxc, float* __restrict__ bc) {
  unsigned i = blockIdx.x * blockDim.x + threadIdx.x;      // 0..4H-1
  unsigned gate = i / Hdim, col = i % Hdim;
  const float* Wx = gate == 0 ? Wgx : gate == 1 ? Wix : gate == 2 ? Wfx : Wox;
  const float* bb = gate == 0 ? bg  : gate == 1 ? bi  : gate == 2 ? bff : bo;
  Wxc[i] = Wx[col];
  bc[i]  = bb[col];
}

__global__ void k_init(unsigned short* __restrict__ h0, float* __restrict__ c) {
  unsigned i = blockIdx.x * blockDim.x + threadIdx.x;
  h0[i] = 0;
  c[i]  = 0.0f;
}

// ---------------------------------------------------------------------------
// One LSTM timestep, fully fused:
//   Z[B,4H] = hPrev @ Wh (+ x_t outer Wx + b); gates; c,h update.
// Block tile: M=64 (batch) x N=64 (hidden cols) x all 4 gates.
// 8 waves = (m32 in {0,1}) x (n16 in {0..3}); each wave holds 2 m-subtiles x
// 4 gate accumulators (8 WMMAs per k-step); gate math is register-local,
// A fragment reused 4x, B fragments 2x.
// K loop: 32 steps of K=32, LDS double-buffered via async copies; fully
// unrolled so accumulators coalesce (no phi copies / hazard NOPs).
// ---------------------------------------------------------------------------
__global__ void __launch_bounds__(256) k_lstm_step(
    const unsigned short* __restrict__ hPrev, unsigned short* __restrict__ hNext,
    float* __restrict__ cState, const unsigned* __restrict__ WhP,
    const float* __restrict__ x, const float* __restrict__ Wxc,
    const float* __restrict__ bc, int t) {
  __shared__ unsigned char Bbuf[2][16384];   // [gate][ntl] x 1KB fragments
  __shared__ unsigned char Abuf[2][4096];    // 64 rows x 64B (64x32 bf16, row-major)

  const int tid  = threadIdx.x;
  const int lane = tid & 31;
  const int wid  = tid >> 5;
  const int m32  = wid & 1;     // which 32-row half of the block tile
  const int n16  = wid >> 1;    // which 16-col tile
  const int blockM = blockIdx.x * 64;

  const char* hPrevB = (const char*)hPrev;
  const int ar = tid >> 2;      // A-copy: row 0..63
  const int ac = tid & 3;       // 16B chunk in row

  // B source: per k-step the needed 16KB [gate][ntl] slice is contiguous at
  // WhP + (kb*16 + blockIdx.y)*16KB and congruent with the LDS buffer layout,
  // so the 4 issues share one address pair and use matched imm offsets.
  const char* bsrc0 = (const char*)WhP + ((size_t)blockIdx.y << 14) + tid * 16;
  const char* asrc0 = hPrevB + (size_t)(blockM + ar) * (Hdim * 2) + ac * 16;

  auto issue_loads = [&](int kb, int buf) {
    const char* bsrc = bsrc0 + ((size_t)kb << 18);       // kb*16*16384
    void* bdst = &Bbuf[buf][tid * 16];
    cp_async16<0>(bdst, bsrc);
    cp_async16<4096>(bdst, bsrc);
    cp_async16<8192>(bdst, bsrc);
    cp_async16<12288>(bdst, bsrc);
    cp_async16<0>(&Abuf[buf][ar * 64 + ac * 16], asrc0 + kb * 64);
  };

  v8f acc[2][4];
#pragma unroll
  for (int mt = 0; mt < 2; ++mt)
#pragma unroll
    for (int g = 0; g < 4; ++g) acc[mt][g] = (v8f){};

  union F { uint4 q[2]; v16bf v; };

  issue_loads(0, 0);
#pragma unroll
  for (int kb = 0; kb < 32; ++kb) {
    const int buf = kb & 1;
    if (kb + 1 < 32) { issue_loads(kb + 1, buf ^ 1); wait_async_5(); }
    else             { wait_async_0(); }
    __syncthreads();

    F af[2];
#pragma unroll
    for (int mt = 0; mt < 2; ++mt) {
      const unsigned char* ab =
          &Abuf[buf][(m32 * 32 + mt * 16 + (lane & 15)) * 64 + ((lane >> 4) ? 32 : 0)];
      af[mt].q[0] = *(const uint4*)ab;          // K 0..7   (or 16..23)
      af[mt].q[1] = *(const uint4*)(ab + 16);   // K 8..15  (or 24..31)
    }
    F bfr[4];
#pragma unroll
    for (int g = 0; g < 4; ++g) {
      const unsigned char* bb = &Bbuf[buf][g * 4096 + n16 * 1024 + lane * 32];
      bfr[g].q[0] = *(const uint4*)bb;
      bfr[g].q[1] = *(const uint4*)(bb + 16);
    }
#pragma unroll
    for (int mt = 0; mt < 2; ++mt)
#pragma unroll
      for (int g = 0; g < 4; ++g)
        acc[mt][g] = wmma_bf16(af[mt].v, bfr[g].v, acc[mt][g]);
    __syncthreads();
  }

  // Epilogue: rank-1 x-term + bias, gates, state update — all in registers.
  const int nG = blockIdx.y * 64 + n16 * 16 + (lane & 15);
  const float wx_g = Wxc[0 * Hdim + nG], wx_i = Wxc[1 * Hdim + nG];
  const float wx_f = Wxc[2 * Hdim + nG], wx_o = Wxc[3 * Hdim + nG];
  const float b_g = bc[0 * Hdim + nG], b_i = bc[1 * Hdim + nG];
  const float b_f = bc[2 * Hdim + nG], b_o = bc[3 * Hdim + nG];
#pragma unroll
  for (int mt = 0; mt < 2; ++mt) {
    const int mBase = blockM + m32 * 32 + mt * 16 + ((lane >> 4) << 3);
#pragma unroll
    for (int v = 0; v < 8; ++v) {
      const int m = mBase + v;
      const float xv = x[(size_t)m * Tdim + t];
      const float zg = acc[mt][0][v] + xv * wx_g + b_g;
      const float zi = acc[mt][1][v] + xv * wx_i + b_i;
      const float zf = acc[mt][2][v] + xv * wx_f + b_f;
      const float zo = acc[mt][3][v] + xv * wx_o + b_o;
      const float g_ = fast_tanh(zg);
      const float i_ = fsigmoid(zi);
      const float f_ = fsigmoid(zf);
      const float o_ = fsigmoid(zo);
      const size_t idx = (size_t)m * Hdim + nG;
      const float cN = g_ * i_ + cState[idx] * f_;
      cState[idx] = cN;
      hNext[idx]  = f2bf(fast_tanh(cN) * o_);
    }
  }
}

// Final projection: out[B,C] = hFinal @ Wph + bp  (bf16 WMMA, direct loads).
__global__ void __launch_bounds__(256) k_proj(const unsigned short* __restrict__ hFin,
                                              const unsigned* __restrict__ WphP,
                                              const float* __restrict__ bp,
                                              float* __restrict__ out) {
  const int tid = threadIdx.x, lane = tid & 31, w = tid >> 5;
  const int mt = blockIdx.x;   // 16 blocks, one 16-row slab each
  const int nt = w;            // 8 waves cover 8 N-tiles of 16
  const char* hB = (const char*)hFin;
  const char* wB = (const char*)WphP;
  v8f acc = {};
#pragma unroll
  for (int kb = 0; kb < 32; ++kb) {
    union { uint4 q[2]; v16bf v; } af, bfr;
    const char* ap = hB + (size_t)(mt * 16 + (lane & 15)) * (Hdim * 2) + kb * 64 +
                     ((lane >> 4) ? 32 : 0);
    af.q[0] = *(const uint4*)ap;
    af.q[1] = *(const uint4*)(ap + 16);
    const char* bpt = wB + (size_t)(kb * 8 + nt) * 1024 + lane * 32;
    bfr.q[0] = *(const uint4*)bpt;
    bfr.q[1] = *(const uint4*)(bpt + 16);
    acc = wmma_bf16(af.v, bfr.v, acc);
  }
  const int n = nt * 16 + (lane & 15);
  const float bias = bp[n];
  const int mBase = mt * 16 + ((lane >> 4) << 3);
#pragma unroll
  for (int v = 0; v < 8; ++v) out[(size_t)(mBase + v) * Cdim + n] = acc[v] + bias;
}

extern "C" void kernel_launch(void* const* d_in, const int* in_sizes, int n_in,
                              void* d_out, int out_size, void* d_ws, size_t ws_size,
                              hipStream_t stream) {
  (void)in_sizes; (void)n_in; (void)out_size; (void)ws_size;
  const float* x   = (const float*)d_in[0];
  const float* Wgx = (const float*)d_in[1];
  const float* Wgh = (const float*)d_in[2];
  const float* Wix = (const float*)d_in[3];
  const float* Wih = (const float*)d_in[4];
  const float* Wfx = (const float*)d_in[5];
  const float* Wfh = (const float*)d_in[6];
  const float* Wox = (const float*)d_in[7];
  const float* Woh = (const float*)d_in[8];
  const float* Wph = (const float*)d_in[9];
  const float* bg  = (const float*)d_in[10];
  const float* bi  = (const float*)d_in[11];
  const float* bf  = (const float*)d_in[12];
  const float* bo  = (const float*)d_in[13];
  const float* bp  = (const float*)d_in[14];

  char* ws = (char*)d_ws;
  unsigned* WhP  = (unsigned*)ws;  ws += (size_t)4 * Hdim * Hdim * 2;  // 8 MB bf16
  unsigned* WphP = (unsigned*)ws;  ws += (size_t)Hdim * Cdim * 2;      // 256 KB
  float* Wxc = (float*)ws;         ws += (size_t)4 * Hdim * 4;
  float* bc  = (float*)ws;         ws += (size_t)4 * Hdim * 4;
  unsigned short* h0 = (unsigned short*)ws; ws += (size_t)Bdim * Hdim * 2;
  unsigned short* h1 = (unsigned short*)ws; ws += (size_t)Bdim * Hdim * 2;
  float* cState = (float*)ws;      ws += (size_t)Bdim * Hdim * 4;

  k_pack_Wh <<<8192, 256, 0, stream>>>(Wgh, Wih, Wfh, Woh, WhP);
  k_pack_Wph<<<256,  256, 0, stream>>>(Wph, WphP);
  k_pack_wxb<<<16,   256, 0, stream>>>(Wgx, Wix, Wfx, Wox, bg, bi, bf, bo, Wxc, bc);
  k_init    <<<(Bdim * Hdim) / 256, 256, 0, stream>>>(h0, cState);

  unsigned short* bufs[2] = {h0, h1};
  dim3 grid(Bdim / 64, Hdim / 64);   // 4 x 16 = 64 blocks per step
  for (int t = 0; t < Tdim; ++t) {
    k_lstm_step<<<grid, 256, 0, stream>>>(bufs[t & 1], bufs[(t + 1) & 1], cState,
                                          WhP, x, Wxc, bc, t);
  }
  k_proj<<<Bdim / 16, 256, 0, stream>>>(bufs[Tdim & 1], WphP, bp, (float*)d_out);
}
